// HeadAttention_609885356198
// MI455X (gfx1250) — matrix operand order repaired and checked by reading, not compiled, not run on hardware
//
#include <hip/hip_runtime.h>
#include <hip/hip_bf16.h>

typedef _Float16 half_t;
typedef __attribute__((ext_vector_type(16))) _Float16 v16h;
typedef __attribute__((ext_vector_type(8)))  _Float16 v8h;
typedef __attribute__((ext_vector_type(8)))  float    v8f;
typedef __attribute__((ext_vector_type(4)))  float    v4f;

#define B_    8
#define T_    2048
#define C_    1024
#define H_    64
#define ROWS  (B_ * T_)          // 16384 rows of x / q / k / v
#define SCALE 0.03125f           // C^-0.5 = 1/32 (reference scales by n_embed^-0.5)
#define NEGINF (-1e30f)
#define KSPLIT 4                 // waves cooperating on one query tile

static __device__ __forceinline__ v8f wmma_f16(v16h a, v16h b, v8f c) {
  // D = A(16x32 f16) * B(32x16 f16) + C(16x16 f32)
  return __builtin_amdgcn_wmma_f32_16x16x32_f16(false, a, false, b, (short)0, c,
                                                false, false);
}

static __device__ __forceinline__ v8f v8f_zero() {
  v8f z;
#pragma unroll
  for (int i = 0; i < 8; ++i) z[i] = 0.0f;
  return z;
}

// ---------------------------------------------------------------------------
// Kernel 0: W [1024,64] f32  ->  Wt [3][64][1024] f16 (transposed, contiguous K)
// ---------------------------------------------------------------------------
__global__ void wt_kernel(const float* __restrict__ Wq, const float* __restrict__ Wk,
                          const float* __restrict__ Wv, half_t* __restrict__ Wt) {
  int tid = blockIdx.x * blockDim.x + threadIdx.x;
  if (tid >= 3 * C_ * H_) return;
  int mat = tid >> 16;            // / 65536
  int rem = tid & 65535;
  int c = rem >> 6;
  int h = rem & 63;
  const float* W = (mat == 0) ? Wq : (mat == 1) ? Wk : Wv;
  Wt[(size_t)mat * H_ * C_ + (size_t)h * C_ + c] = (half_t)W[(size_t)c * H_ + h];
}

// ---------------------------------------------------------------------------
// Kernel 1: projections. One wave per (16-row tile, matrix).
// Q,K stored row-major f16 [16384,64]; V stored TRANSPOSED f16 [64,16384]
// so the attention P*V B-fragment is a contiguous 32B load.
// ---------------------------------------------------------------------------
__global__ __launch_bounds__(256) void proj_kernel(const float* __restrict__ x,
                                                   const half_t* __restrict__ Wt,
                                                   half_t* __restrict__ qkv) {
  int lane = threadIdx.x & 31;
  int wave = threadIdx.x >> 5;
  int task = blockIdx.x * 8 + wave;   // 0..3071, exact
  int mat  = task % 3;
  int rt   = task / 3;                // 0..1023 (16-row tiles over B*T)

  int colh = lane & 15;
  int b0   = (lane & 16) ? 8 : 0;     // unused; A frag also splits K by klo
  int klo  = (lane & 16) ? 16 : 0;    // B-frag K split
  int hi8  = (lane & 16) ? 8 : 0;     // C/D row split

  const float*  xrow = x  + (size_t)(rt * 16 + colh) * C_;   // A row = lane&15
  const half_t* wb   = Wt + (size_t)mat * H_ * C_;

  v8f acc[4];
#pragma unroll
  for (int c = 0; c < 4; ++c) acc[c] = v8f_zero();

  for (int k0 = 0; k0 < C_; k0 += 32) {
    __builtin_prefetch(xrow + k0 + 128, 0, 0);   // global_prefetch_b8
    // Build A fragment (f32 -> f16), contiguous K: lane half h holds k0+16h..k0+16h+15.
    const float* p = xrow + k0 + klo;
    v4f f0 = *(const v4f*)(p);
    v4f f1 = *(const v4f*)(p + 4);
    v4f f2 = *(const v4f*)(p + 8);
    v4f f3 = *(const v4f*)(p + 12);
    v16h a;
#pragma unroll
    for (int i = 0; i < 4; ++i) {
      a[i]      = (half_t)f0[i];
      a[4 + i]  = (half_t)f1[i];
      a[8 + i]  = (half_t)f2[i];
      a[12 + i] = (half_t)f3[i];
    }
#pragma unroll
    for (int c = 0; c < 4; ++c) {
      const half_t* colp = wb + (size_t)(c * 16 + colh) * C_ + k0 + klo;
      v16h bfrag = *(const v16h*)colp;           // 32B contiguous load
      acc[c] = wmma_f16(a, bfrag, acc[c]);
    }
  }

  if (mat == 2) {
    // V transposed: Vt[h][rowglobal], rows consecutive per lane -> packed stores
    half_t* vt = qkv + (size_t)2 * ROWS * H_;
#pragma unroll
    for (int c = 0; c < 4; ++c) {
      v8h pk;
#pragma unroll
      for (int j = 0; j < 8; ++j) pk[j] = (half_t)acc[c][j];
      half_t* dst = vt + (size_t)(c * 16 + colh) * ROWS + rt * 16 + hi8;
      *(v8h*)dst = pk;                           // 16B contiguous store
    }
  } else {
    half_t* out = qkv + (size_t)mat * ROWS * H_;
#pragma unroll
    for (int c = 0; c < 4; ++c)
#pragma unroll
      for (int j = 0; j < 8; ++j)
        out[(size_t)(rt * 16 + j + hi8) * H_ + c * 16 + colh] = (half_t)acc[c][j];
  }
}

// ---------------------------------------------------------------------------
// Kernel 2: causal flash attention, 4-way key-split.
// Block = 128 threads = 4 waves = one (batch, 16-query tile).
// Wave w handles key chunks ch == w (mod 4); partials merged via LDS (LSE combine).
// ---------------------------------------------------------------------------
__global__ __launch_bounds__(128) void attn_kernel(const half_t* __restrict__ qkv,
                                                   float* __restrict__ out) {
  __shared__ float  Osh[KSPLIT][16][H_];     // 16 KB partial O
  __shared__ float  msh[KSPLIT][16], lsh[KSPLIT][16];
  __shared__ float  ew[KSPLIT][16], linv[16];
  __shared__ half_t plds[KSPLIT][16 * 32];   // per-wave P relayout tile

  int lane = threadIdx.x & 31;
  int w    = threadIdx.x >> 5;          // ksplit wave id 0..3
  int task = blockIdx.x;                // 0..1023
  int b    = task >> 7;                 // batch
  int q0   = (task & 127) * 16;         // query tile base

  const half_t* Qb = qkv + ((size_t)b * T_) * H_;
  const half_t* Kb = qkv + (size_t)ROWS * H_ + ((size_t)b * T_) * H_;
  const half_t* Vt = qkv + (size_t)2 * ROWS * H_;   // [64][16384]

  int col = lane & 15;
  int b0  = (lane & 16) ? 8 : 0;
  int klo = (lane & 16) ? 16 : 0;
  int hi8 = (lane & 16) ? 8 : 0;

  // Q A-fragments for k0 = 0, 32 (H = 64)
  const half_t* qrow = Qb + (size_t)(q0 + col) * H_;
  v16h qa[2];
#pragma unroll
  for (int t = 0; t < 2; ++t) {
    v8h lo = *(const v8h*)(qrow + t * 32 + klo);
    v8h hi = *(const v8h*)(qrow + t * 32 + klo + 8);
#pragma unroll
    for (int i = 0; i < 8; ++i) { qa[t][i] = lo[i]; qa[t][8 + i] = hi[i]; }
  }

  v8f o[4];
#pragma unroll
  for (int c = 0; c < 4; ++c) o[c] = v8f_zero();
  float m[8], l[8];
#pragma unroll
  for (int j = 0; j < 8; ++j) { m[j] = NEGINF; l[j] = 0.0f; }

  int nch = (q0 + 15) / 32 + 1;   // 32-key chunks covering keys <= q0+15
  for (int ch = w; ch < nch; ch += KSPLIT) {
    int s0 = ch * 32;

    // ---- S = Q K^T for two 16-key tiles ----
    v8f s[2];
#pragma unroll
    for (int t = 0; t < 2; ++t) {
      const half_t* krow = Kb + (size_t)(s0 + t * 16 + col) * H_;
      v16h kb0 = *(const v16h*)(krow + klo);        // h = 0..31
      v16h kb1 = *(const v16h*)(krow + 32 + klo);   // h = 32..63
      v8f sc = v8f_zero();
      sc = wmma_f16(qa[0], kb0, sc);
      sc = wmma_f16(qa[1], kb1, sc);
      s[t] = sc;
    }

    // ---- scale + causal mask ----
    float sc0[8], sc1[8];
#pragma unroll
    for (int j = 0; j < 8; ++j) {
      int r = q0 + j + hi8;
      float a0 = s[0][j] * SCALE;
      float a1 = s[1][j] * SCALE;
      if (s0 + col > r)      a0 = NEGINF;
      if (s0 + 16 + col > r) a1 = NEGINF;
      sc0[j] = a0; sc1[j] = a1;
    }

    // ---- chunk row max (butterfly within the 16-lane half of each row) ----
    float mc[8];
#pragma unroll
    for (int j = 0; j < 8; ++j) mc[j] = fmaxf(sc0[j], sc1[j]);
#pragma unroll
    for (int msk = 1; msk < 16; msk <<= 1)
#pragma unroll
      for (int j = 0; j < 8; ++j) mc[j] = fmaxf(mc[j], __shfl_xor(mc[j], msk, 32));

    float f[8];
#pragma unroll
    for (int j = 0; j < 8; ++j) {
      float mn = fmaxf(m[j], mc[j]);
      f[j] = __expf(m[j] - mn);
      m[j] = mn;
    }
    float p0[8], p1[8], rs[8];
#pragma unroll
    for (int j = 0; j < 8; ++j) {
      p0[j] = __expf(sc0[j] - m[j]);
      p1[j] = __expf(sc1[j] - m[j]);
      rs[j] = p0[j] + p1[j];
    }
#pragma unroll
    for (int msk = 1; msk < 16; msk <<= 1)
#pragma unroll
      for (int j = 0; j < 8; ++j) rs[j] += __shfl_xor(rs[j], msk, 32);
#pragma unroll
    for (int j = 0; j < 8; ++j) l[j] = l[j] * f[j] + rs[j];
#pragma unroll
    for (int c = 0; c < 4; ++c)
#pragma unroll
      for (int j = 0; j < 8; ++j) o[c][j] *= f[j];

    // ---- relayout P (C-layout) -> A-fragment via per-wave LDS tile ----
    half_t* pw = &plds[w][0];
#pragma unroll
    for (int j = 0; j < 8; ++j) {
      int row = j + hi8;
      pw[row * 32 + col]      = (half_t)p0[j];
      pw[row * 32 + 16 + col] = (half_t)p1[j];
    }
    __builtin_amdgcn_wave_barrier();   // keep in-order DS store->load per wave
    v16h pa;
    {
      const half_t* pr = pw + (size_t)col * 32;   // A row = lane & 15
      v8h lo = *(const v8h*)(pr + klo);
      v8h hi = *(const v8h*)(pr + klo + 8);
#pragma unroll
      for (int i = 0; i < 8; ++i) { pa[i] = lo[i]; pa[8 + i] = hi[i]; }
    }

    // ---- O += P (16x32) * V (32x64); Vt gives contiguous 32B B-fragments ----
#pragma unroll
    for (int c = 0; c < 4; ++c) {
      const half_t* vp = Vt + (size_t)(c * 16 + col) * ROWS + (size_t)b * T_ + s0 + klo;
      v16h vb = *(const v16h*)vp;                // one 32B load (was 16 scalars)
      o[c] = wmma_f16(pa, vb, o[c]);
    }
  }

  // ---- publish per-wave partials ----
#pragma unroll
  for (int c = 0; c < 4; ++c)
#pragma unroll
    for (int j = 0; j < 8; ++j)
      Osh[w][j + hi8][c * 16 + col] = o[c][j];
  if (col == 0) {                         // lanes 0 and 16 hold rows 0-7 / 8-15
#pragma unroll
    for (int j = 0; j < 8; ++j) { msh[w][j + hi8] = m[j]; lsh[w][j + hi8] = l[j]; }
  }
  __syncthreads();

  // ---- merge factors: one thread per row ----
  if (threadIdx.x < 16) {
    int r = threadIdx.x;
    float mg = msh[0][r];
#pragma unroll
    for (int ww = 1; ww < KSPLIT; ++ww) mg = fmaxf(mg, msh[ww][r]);
    float sum = 0.0f;
#pragma unroll
    for (int ww = 0; ww < KSPLIT; ++ww) {
      float e = __expf(msh[ww][r] - mg);
      ew[ww][r] = e;
      sum += lsh[ww][r] * e;
    }
    linv[r] = 1.0f / sum;
  }
  __syncthreads();

  // ---- weighted combine + store: 128 threads x 8 elements ----
  float* ob = out + ((size_t)b * T_ + q0) * H_;
  for (int e = threadIdx.x; e < 16 * H_; e += 128) {
    int r  = e >> 6;
    int cc = e & 63;
    float acc = 0.0f;
#pragma unroll
    for (int ww = 0; ww < KSPLIT; ++ww) acc += Osh[ww][r][cc] * ew[ww][r];
    ob[(size_t)r * H_ + cc] = acc * linv[r];
  }
}

// ---------------------------------------------------------------------------
extern "C" void kernel_launch(void* const* d_in, const int* in_sizes, int n_in,
                              void* d_out, int out_size, void* d_ws, size_t ws_size,
                              hipStream_t stream) {
  const float* x  = (const float*)d_in[0];
  const float* Wq = (const float*)d_in[1];
  const float* Wk = (const float*)d_in[2];
  const float* Wv = (const float*)d_in[3];
  float* out = (float*)d_out;

  half_t* ws  = (half_t*)d_ws;
  half_t* Wt  = ws;                          // 3*64*1024 halves   (384 KB)
  half_t* qkv = ws + 3 * H_ * C_;            // 3*16384*64 halves  (6 MB)

  wt_kernel<<<(3 * C_ * H_ + 255) / 256, 256, 0, stream>>>(Wq, Wk, Wv, Wt);
  proj_kernel<<<384, 256, 0, stream>>>(x, Wt, qkv);   // 3072 wave-tasks
  attn_kernel<<<1024, 128, 0, stream>>>(qkv, out);    // 4096 wave-tasks
}
